// MemoryBank_82480551952470
// MI455X (gfx1250) — matrix operand order repaired and checked by reading, not compiled, not run on hardware
//
#include <hip/hip_runtime.h>
#include <hip/hip_bf16.h>

// ---------------------------------------------------------------------------
// MemoryBank forward for MI455X (gfx1250, wave32, WMMA 16x16x32 bf16)
// B=4, S=2048, D=1024, N=512, H=8, HD=128  ->  ROWS = B*S = 8192
// ---------------------------------------------------------------------------

#define ROWS   8192
#define DIM    1024
#define NMEM   512
#define K2DIM  2048
#define LN_EPS 1e-5f
#define SCALE_QK 0.08838834764831845f   // 1/sqrt(128)

typedef __bf16 bf16_t;
typedef __attribute__((ext_vector_type(16))) __bf16 v16bf;
typedef __attribute__((ext_vector_type(8)))  float  v8f;

union FragBF {
    v16bf v;
    uint4 q[2];
    unsigned short u[16];
};

__device__ __forceinline__ unsigned short f2bf(float f) {
    unsigned int u = __float_as_uint(f);
    u += 0x7FFFu + ((u >> 16) & 1u);      // round-to-nearest-even
    return (unsigned short)(u >> 16);
}

// A-matrix (16x32 bf16, MxK) per-lane packed offset for element (m, kk):
// lanes 0-15: M=lane, K in {0..7, 16..23}; lanes 16-31: M=lane-16, K in {8..15, 24..31}
__device__ __forceinline__ int packA_off(int m, int kk) {
    int half = (kk >> 3) & 1;
    int v    = ((kk >> 4) << 2) + ((kk & 7) >> 1);
    int pos  = kk & 1;
    return ((m + (half << 4)) << 4) + (v << 1) + pos;
}

// B-matrix (32x16 bf16, KxN) per-lane packed offset for element (kk, nn):
// lanes 0-15: K in {0..7, 16..23} (col N=lane), lanes 16-31: K in {8..15, 24..31} (col N=lane-16)
__device__ __forceinline__ int packB_off(int kk, int nn) {
    int lane = nn + (((kk >> 3) & 1) << 4);
    int v    = ((kk >> 4) << 2) + ((kk & 7) >> 1);
    int pos  = kk & 1;
    return (lane << 4) + (v << 1) + pos;
}

__device__ __forceinline__ void load_frag(FragBF& f, const unsigned short* p) {
    const uint4* p4 = (const uint4*)p;    // 32B contiguous per lane -> 2x b128
    f.q[0] = p4[0];
    f.q[1] = p4[1];
}

__device__ __forceinline__ v8f wmma_bf16(const FragBF& a, const FragBF& b, v8f c) {
    return __builtin_amdgcn_wmma_f32_16x16x32_bf16(false, a.v, false, b.v,
                                                   (short)0, c, false, false);
}

// CDNA5 async global->LDS copy: one instruction moves 16B per lane (512B/wave)
// directly into LDS, tracked by ASYNCcnt.  lds_off = byte offset in LDS
// (low 32 bits of the generic shared-memory address), g = 64-bit global addr.
__device__ __forceinline__ void async_copy_b128(unsigned int lds_off,
                                                const void* g) {
    asm volatile("global_load_async_to_lds_b128 %0, %1, off"
                 :: "v"(lds_off), "v"((unsigned long long)g)
                 : "memory");
}
__device__ __forceinline__ void async_wait0() {
    asm volatile("s_wait_asynccnt 0x0" ::: "memory");
}

// ---------------------------------------------------------------------------
// Packing kernels (run once per launch; weights are tiny, L2-resident)
// ---------------------------------------------------------------------------

// Generic: src is row-major [K][Ncols] f32 -> WMMA-B packed bf16
__global__ void pack_b_kernel(const float* __restrict__ src,
                              unsigned short* __restrict__ dst,
                              int K, int Ncols) {
    int total = K * Ncols;
    int NT = Ncols >> 4;
    for (int i = blockIdx.x * blockDim.x + threadIdx.x; i < total;
         i += gridDim.x * blockDim.x) {
        int k = i / Ncols, n = i - k * Ncols;
        int kt = k >> 5, kk = k & 31, nt = n >> 4, nn = n & 15;
        dst[(((kt * NT) + nt) << 9) + packB_off(kk, nn)] = f2bf(src[i]);
    }
}

// memory_keys [N][D] -> B matrix for Q*K^T:  K-dim = D, cols = N (transpose)
__global__ void pack_keysT_kernel(const float* __restrict__ keys,
                                  unsigned short* __restrict__ dst) {
    for (int i = blockIdx.x * blockDim.x + threadIdx.x; i < NMEM * DIM;
         i += gridDim.x * blockDim.x) {
        int n = i >> 10, d = i & 1023;                 // coalesced read over d
        int kt = d >> 5, kk = d & 31, nt = n >> 4, nn = n & 15;
        dst[(((kt << 5) + nt) << 9) + packB_off(kk, nn)] = f2bf(keys[i]);
    }
}

// update_input [8192][1024] -> second K-half (kt 32..63) of packed gate-A
__global__ void pack_update_kernel(const float* __restrict__ u,
                                   unsigned short* __restrict__ gateA) {
    for (int i = blockIdx.x * blockDim.x + threadIdx.x; i < ROWS * DIM;
         i += gridDim.x * blockDim.x) {
        int row = i >> 10, d = i & 1023;
        int rt = row >> 4, m = row & 15;
        int kt = 32 + (d >> 5), kk = d & 31;
        gateA[(((rt << 6) + kt) << 9) + packA_off(m, kk)] = f2bf(u[i]);
    }
}

// ---------------------------------------------------------------------------
// Kernel 1: attention (scores -> masked softmax -> P*V*decay -> packed gate-A)
// One block per 16-row tile; 8 waves.
// ---------------------------------------------------------------------------
__global__ __launch_bounds__(256) void attn_kernel(
    const float* __restrict__ query, const float* __restrict__ mw,
    const unsigned short* __restrict__ keysT,   // packed B, K=1024, N=512
    const unsigned short* __restrict__ valsB,   // packed B, K=512,  N=1024
    const float* __restrict__ decay_p,
    unsigned short* __restrict__ gateA) {
    __shared__ alignas(16) unsigned short qA[16 * DIM];    // 32 KB, A-packed Q
    __shared__ alignas(16) float          sc[16 * NMEM];   // 32 KB, scores
    __shared__ alignas(16) unsigned short pA[16 * NMEM];   // 16 KB, A-packed P
    __shared__ alignas(16) float          mwS[16 * NMEM];  // 32 KB, weights

    const int tid = threadIdx.x, wave = tid >> 5, lane = tid & 31;
    const int rt = blockIdx.x;
    const long rowBase = (long)rt * 16;

    // --- coalesced prestage of the memory_weights tile ----------------------
    {
        const float* msrc = mw + rowBase * NMEM;
        for (int i = tid; i < 16 * NMEM; i += 256) mwS[i] = msrc[i];
    }
    // --- load + pack Q tile into A-fragment layout --------------------------
    const float* qsrc = query + rowBase * DIM;
    for (int i = tid; i < 16 * DIM; i += 256) {
        int m = i >> 10, k = i & 1023;
        qA[((k >> 5) << 9) + packA_off(m, k & 31)] = f2bf(qsrc[i]);
    }
    __syncthreads();

    // --- scores = Q*K^T : each wave owns 4 key blocks (32 total) ------------
    {
        v8f acc[4];
        for (int c = 0; c < 4; ++c)
            for (int r = 0; r < 8; ++r) acc[c][r] = 0.0f;
        for (int kt = 0; kt < 32; ++kt) {
            FragBF a; load_frag(a, &qA[(kt << 9) + (lane << 4)]);
            for (int c = 0; c < 4; ++c) {
                int kb = (wave << 2) + c;
                FragBF b; load_frag(b, keysT + ((((kt << 5) + kb) << 9) + (lane << 4)));
                acc[c] = wmma_bf16(a, b, acc[c]);
            }
        }
        int n = lane & 15, mB = (lane >> 4) << 3;
        for (int c = 0; c < 4; ++c) {
            int col = (((wave << 2) + c) << 4) + n;
            for (int r = 0; r < 8; ++r) {
                int m = mB + r;
                sc[(m << 9) + col] = acc[c][r] * SCALE_QK * mwS[(m << 9) + col];
            }
        }
    }
    __syncthreads();

    // --- softmax over N=512 : each wave handles 2 rows ----------------------
    for (int rr = 0; rr < 2; ++rr) {
        int m = (wave << 1) + rr;
        float x[16];
        float mx = -3.4e38f;
        for (int i = 0; i < 16; ++i) {
            x[i] = sc[(m << 9) + lane + (i << 5)];
            mx = fmaxf(mx, x[i]);
        }
        for (int off = 16; off >= 1; off >>= 1) mx = fmaxf(mx, __shfl_xor(mx, off, 32));
        float s = 0.0f;
        for (int i = 0; i < 16; ++i) { x[i] = __expf(x[i] - mx); s += x[i]; }
        for (int off = 16; off >= 1; off >>= 1) s += __shfl_xor(s, off, 32);
        float inv = 1.0f / s;
        for (int i = 0; i < 16; ++i)            // col = lane + 32*i -> kk=lane, kt=i
            pA[(i << 9) + packA_off(m, lane)] = f2bf(x[i] * inv);
    }
    __syncthreads();

    // --- R = P*V * decay -> write into first K-half of packed gate-A --------
    {
        const float decay = decay_p[0];
        v8f acc[8];
        for (int c = 0; c < 8; ++c)
            for (int r = 0; r < 8; ++r) acc[c][r] = 0.0f;
        for (int kt = 0; kt < 16; ++kt) {
            FragBF a; load_frag(a, &pA[(kt << 9) + (lane << 4)]);
            for (int c = 0; c < 8; ++c) {
                int cb = (wave << 3) + c;
                FragBF b; load_frag(b, valsB + ((((kt << 6) + cb) << 9) + (lane << 4)));
                acc[c] = wmma_bf16(a, b, acc[c]);
            }
        }
        int n = lane & 15, mB = (lane >> 4) << 3;
        for (int c = 0; c < 8; ++c) {
            int col = (((wave << 3) + c) << 4) + n;
            int kt2 = col >> 5, kk2 = col & 31;
            for (int r = 0; r < 8; ++r) {
                int m = mB + r;
                gateA[(((rt << 6) + kt2) << 9) + packA_off(m, kk2)] =
                    f2bf(acc[c][r] * decay);
            }
        }
    }
}

// ---------------------------------------------------------------------------
// Kernel 2: gate GEMM [8192x2048]@[2048x1024] + sigmoid + row mean
// A-strip (64 KB, contiguous in packed layout) async-staged into LDS once per
// block -> inner-loop A loads are ds_load_b128, no 8x-redundant global reads.
// ---------------------------------------------------------------------------
__global__ __launch_bounds__(256) void gate_kernel(
    const unsigned short* __restrict__ gateA,   // packed A, K=2048
    const unsigned short* __restrict__ gw,      // packed B, K=2048, N=1024
    const float* __restrict__ gb,
    float* __restrict__ rowscale) {
    __shared__ alignas(16) unsigned short aS[64 * 512];   // 64 KB A strip
    __shared__ float rowsum[16];
    const int tid = threadIdx.x, wave = tid >> 5, lane = tid & 31;
    const int rt = blockIdx.x;
    if (tid < 16) rowsum[tid] = 0.0f;

    // async global->LDS stage of this block's packed A strip (contiguous)
    {
        const unsigned short* strip = gateA + ((long)rt << 15);   // 32768 elems
        unsigned int ldsbase = (unsigned int)(unsigned long long)(void*)aS;
        for (int it = tid; it < 4096; it += 256)                  // 4096 x 16B
            async_copy_b128(ldsbase + it * 16, (const char*)strip + it * 16);
        async_wait0();
    }
    __syncthreads();

    v8f acc[8];
    for (int c = 0; c < 8; ++c)
        for (int r = 0; r < 8; ++r) acc[c][r] = 0.0f;
    for (int kt = 0; kt < 64; ++kt) {
        FragBF a; load_frag(a, &aS[(kt << 9) + (lane << 4)]);
        for (int c = 0; c < 8; ++c) {
            int cb = (wave << 3) + c;
            FragBF b; load_frag(b, gw + ((((kt << 6) + cb) << 9) + (lane << 4)));
            acc[c] = wmma_bf16(a, b, acc[c]);
        }
    }
    int n = lane & 15, mB = (lane >> 4) << 3;
    float part[8];
    for (int r = 0; r < 8; ++r) part[r] = 0.0f;
    for (int c = 0; c < 8; ++c) {
        int col = (((wave << 3) + c) << 4) + n;
        float bias = gb[col];
        for (int r = 0; r < 8; ++r) {
            float z = acc[c][r] + bias;
            part[r] += 1.0f / (1.0f + __expf(-z));
        }
    }
    for (int r = 0; r < 8; ++r) atomicAdd(&rowsum[mB + r], part[r]);
    __syncthreads();
    if (tid < 16)
        rowscale[rt * 16 + tid] = 1.0f + rowsum[tid] * (1.0f / (float)DIM);
}

// ---------------------------------------------------------------------------
// Kernel 3: out = rowscale*(R @ out_w) + out_b, then LayerNorm over D
// ---------------------------------------------------------------------------
__global__ __launch_bounds__(256) void out_kernel(
    const unsigned short* __restrict__ gateA,   // first K-half = packed R
    const unsigned short* __restrict__ ow,      // packed B, K=1024, N=1024
    const float* __restrict__ ob, const float* __restrict__ rowscale,
    const float* __restrict__ gamma, const float* __restrict__ beta,
    float* __restrict__ out) {
    __shared__ alignas(16) unsigned short aS[32 * 512];   // 32 KB A strip (R)
    __shared__ alignas(16) float ot[16 * DIM];            // 64 KB
    __shared__ float rs[16];
    const int tid = threadIdx.x, wave = tid >> 5, lane = tid & 31;
    const int rt = blockIdx.x;
    if (tid < 16) rs[tid] = rowscale[rt * 16 + tid];

    // async global->LDS stage of the R half (kt 0..31) of this block's strip
    {
        const unsigned short* strip = gateA + ((long)rt << 15);   // R = first 32KB
        unsigned int ldsbase = (unsigned int)(unsigned long long)(void*)aS;
        for (int it = tid; it < 2048; it += 256)                  // 2048 x 16B
            async_copy_b128(ldsbase + it * 16, (const char*)strip + it * 16);
        async_wait0();
    }
    __syncthreads();

    v8f acc[8];
    for (int c = 0; c < 8; ++c)
        for (int r = 0; r < 8; ++r) acc[c][r] = 0.0f;
    for (int kt = 0; kt < 32; ++kt) {
        FragBF a; load_frag(a, &aS[(kt << 9) + (lane << 4)]);
        for (int c = 0; c < 8; ++c) {
            int cb = (wave << 3) + c;
            FragBF b; load_frag(b, ow + ((((kt << 6) + cb) << 9) + (lane << 4)));
            acc[c] = wmma_bf16(a, b, acc[c]);
        }
    }
    int n = lane & 15, mB = (lane >> 4) << 3;
    for (int c = 0; c < 8; ++c) {
        int col = (((wave << 3) + c) << 4) + n;
        float bias = ob[col];
        for (int r = 0; r < 8; ++r) {
            int m = mB + r;
            ot[(m << 10) + col] = acc[c][r] * rs[m] + bias;
        }
    }
    __syncthreads();

    // LayerNorm: each wave handles 2 rows
    for (int rr = 0; rr < 2; ++rr) {
        int m = (wave << 1) + rr;
        float x[32], s = 0.0f, s2 = 0.0f;
        for (int i = 0; i < 32; ++i) {
            float v = ot[(m << 10) + lane + (i << 5)];
            x[i] = v; s += v; s2 += v * v;
        }
        for (int off = 16; off >= 1; off >>= 1) {
            s  += __shfl_xor(s,  off, 32);
            s2 += __shfl_xor(s2, off, 32);
        }
        float mean = s * (1.0f / (float)DIM);
        float var  = s2 * (1.0f / (float)DIM) - mean * mean;
        float rstd = rsqrtf(var + LN_EPS);
        long row = (long)rt * 16 + m;
        for (int i = 0; i < 32; ++i) {
            int col = lane + (i << 5);
            out[row * DIM + col] = (x[i] - mean) * rstd * gamma[col] + beta[col];
        }
    }
}

// ---------------------------------------------------------------------------
// Launcher
// ---------------------------------------------------------------------------
extern "C" void kernel_launch(void* const* d_in, const int* in_sizes, int n_in,
                              void* d_out, int out_size, void* d_ws, size_t ws_size,
                              hipStream_t stream) {
    const float* query = (const float*)d_in[0];
    const float* mw    = (const float*)d_in[1];
    const float* upd   = (const float*)d_in[2];
    const float* keys  = (const float*)d_in[3];
    const float* vals  = (const float*)d_in[4];
    const float* decay = (const float*)d_in[5];
    const float* gw    = (const float*)d_in[6];
    const float* gb    = (const float*)d_in[7];
    const float* ow    = (const float*)d_in[8];
    const float* ob    = (const float*)d_in[9];
    const float* gam   = (const float*)d_in[10];
    const float* bet   = (const float*)d_in[11];
    float* out = (float*)d_out;

    // workspace layout (bf16 elements unless noted)
    unsigned short* keysT_p = (unsigned short*)d_ws;        //  524288  (1 MB)
    unsigned short* vals_p  = keysT_p + 524288;             //  524288  (1 MB)
    unsigned short* gw_p    = vals_p  + 524288;             // 2097152  (4 MB)
    unsigned short* ow_p    = gw_p    + 2097152;            // 1048576  (2 MB)
    unsigned short* gateA   = ow_p    + 1048576;            // 16777216 (32 MB)
    float*          rscale  = (float*)(gateA + 16777216);   // 8192 f32

    pack_keysT_kernel<<<1024, 256, 0, stream>>>(keys, keysT_p);
    pack_b_kernel<<<1024, 256, 0, stream>>>(vals, vals_p, NMEM, DIM);
    pack_b_kernel<<<2048, 256, 0, stream>>>(gw, gw_p, K2DIM, DIM);
    pack_b_kernel<<<1024, 256, 0, stream>>>(ow, ow_p, DIM, DIM);
    pack_update_kernel<<<4096, 256, 0, stream>>>(upd, gateA);

    attn_kernel<<<ROWS / 16, 256, 0, stream>>>(query, mw, keysT_p, vals_p, decay, gateA);
    gate_kernel<<<ROWS / 16, 256, 0, stream>>>(gateA, gw_p, gb, rscale);
    out_kernel<<<ROWS / 16, 256, 0, stream>>>(gateA, ow_p, ob, rscale, gam, bet, out);
}
